// MergedMixtralSparseMoeBlock_85117661872396
// MI455X (gfx1250) — matrix operand order repaired and checked by reading, not compiled, not run on hardware
//
#include <hip/hip_runtime.h>

// ---------------- problem constants ----------------
#define TB 2
#define TS 2048
#define T  (TB * TS)          // 4096 tokens
#define H  1024
#define F  2048
#define E  8
#define TOPK 2
#define EFH (E * F * H)       // 16,777,216
#define NROWS (T * TOPK)      // 8192 compact expert rows (sum of counts == NROWS exactly)

// ---------------- WMMA types ----------------
typedef __attribute__((ext_vector_type(16))) __bf16 v16bf;
typedef __attribute__((ext_vector_type(8)))  float  v8f;

union Frag16 {
    unsigned int u[8];
    v16bf v;
};

static __device__ __forceinline__ v8f wmma_bf16(const v16bf a, const v16bf b, const v8f c) {
    // D = A(16x32 bf16) * B(32x16 bf16) + C(16x16 f32)
    return __builtin_amdgcn_wmma_f32_16x16x32_bf16(
        /*neg_a=*/false, a, /*neg_b=*/false, b,
        /*c_mod=*/(short)0, c, /*reuse_a=*/false, /*reuse_b=*/false);
}

static __device__ __forceinline__ unsigned short f32_to_bf16_rne(float f) {
    unsigned int u = __float_as_uint(f);
    u += 0x7FFFu + ((u >> 16) & 1u);   // round-to-nearest-even
    return (unsigned short)(u >> 16);
}

static __device__ __forceinline__ float silu_gate(float h1, float h3) {
    // silu(h1) * h3 with fast v_rcp (result is rounded to bf16 anyway)
    return h1 * __builtin_amdgcn_rcpf(1.f + __expf(-h1)) * h3;
}

static __device__ __forceinline__ unsigned int ld_u32(const unsigned short* p) {
    return *(const unsigned int*)p;    // packed pair of bf16
}

// Load A fragment (16x32 bf16, M = lane&15): K = 2*(v&3) + 16*(v>=4) + 8*half
static __device__ __forceinline__ void load_a(Frag16& a, const unsigned short* row, int k0, int half) {
    #pragma unroll
    for (int v = 0; v < 8; ++v) {
        const int Ka = ((v & 3) * 2) + ((v >> 2) * 16) + half * 8;
        a.u[v] = ld_u32(row + k0 + Ka);
    }
}
// Load B fragment (32x16 bf16, N = lane&15): K = 2*(v&3) + 16*(v>=4) + 8*half
static __device__ __forceinline__ void load_b(Frag16& b, const unsigned short* row, int k0, int half) {
    #pragma unroll
    for (int v = 0; v < 8; ++v) {
        b.u[v] = ld_u32(row + k0 + ((v & 3) * 2) + ((v >> 2) * 16) + half * 8);
    }
}

// ---------------- f32 -> bf16 conversion ----------------
__global__ __launch_bounds__(256) void cvt_bf16_kernel(const float* __restrict__ in,
                                                       unsigned short* __restrict__ out,
                                                       int n) {
    int i = blockIdx.x * blockDim.x + threadIdx.x;
    int stride = gridDim.x * blockDim.x;
    for (; i < n; i += stride) out[i] = f32_to_bf16_rne(in[i]);
}

// ---------------- router: exact f32 logits, top-2, gather + inverse map ----------------
__global__ __launch_bounds__(128) void router_kernel(const float* __restrict__ x,
                                                     const float* __restrict__ gw,
                                                     float* __restrict__ logits_out,
                                                     int* __restrict__ counts,
                                                     int* __restrict__ tok_id,
                                                     int* __restrict__ tok_e2,
                                                     int* __restrict__ tok_pos2,
                                                     float* __restrict__ tok_wt2) {
    const int tok = blockIdx.x;
    const int t   = threadIdx.x;           // 128 threads
    const int e   = t & 7;
    const int s16 = t >> 3;                // 0..15

    __shared__ float part[E][16];
    __shared__ float lg[E];

    float sum = 0.f;
    const float* xr = x + (size_t)tok * H;
    const float* gr = gw + (size_t)e * H;
    for (int h = s16; h < H; h += 16) sum += xr[h] * gr[h];
    part[e][s16] = sum;
    __syncthreads();

    if (t < E) {
        float l = 0.f;
        #pragma unroll
        for (int i = 0; i < 16; ++i) l += part[t][i];
        lg[t] = l;
        logits_out[(size_t)tok * E + t] = l;
    }
    __syncthreads();

    if (t == 0) {
        // top-1 / top-2 with strict > so the first index wins ties (lax.top_k semantics)
        int i1 = 0; float l1 = lg[0];
        #pragma unroll
        for (int i = 1; i < E; ++i) if (lg[i] > l1) { l1 = lg[i]; i1 = i; }
        int i2 = (i1 == 0) ? 1 : 0; float l2 = lg[i2];
        #pragma unroll
        for (int i = 0; i < E; ++i) if (i != i1 && lg[i] > l2) { l2 = lg[i]; i2 = i; }
        // renormalized softmax weights over the two selected logits
        float wa = 1.f / (1.f + __expf(l2 - l1));
        float wb = 1.f - wa;
        int p1 = atomicAdd(&counts[i1], 1);
        tok_id[i1 * T + p1] = tok;
        tok_e2[2 * tok] = i1; tok_pos2[2 * tok] = p1; tok_wt2[2 * tok] = wa;
        int p2 = atomicAdd(&counts[i2], 1);
        tok_id[i2 * T + p2] = tok;
        tok_e2[2 * tok + 1] = i2; tok_pos2[2 * tok + 1] = p2; tok_wt2[2 * tok + 1] = wb;
    }
}

// ---------------- prefix over expert counts -> compact row bases ----------------
__global__ void prefix_kernel(const int* __restrict__ counts, int* __restrict__ base) {
    if (threadIdx.x == 0 && blockIdx.x == 0) {
        int s = 0;
        for (int e = 0; e < E; ++e) { base[e] = s; s += counts[e]; }
    }
}

// ---------------- stage A: gated[row, f] = silu(x@w1^T) * (x@w3^T), bf16 ----------------
// wave tile: 32 tokens x 32 f (2 A frags x 2 B frags per weight -> 8 WMMA / 6KB loads)
// grid: (T/32, F/256, E); block: 256 (8 waves)
__global__ __launch_bounds__(256) void moe_stage1_kernel(const unsigned short* __restrict__ xb,
                                                         const unsigned short* __restrict__ w1b,
                                                         const unsigned short* __restrict__ w3b,
                                                         unsigned short* __restrict__ gated,
                                                         const int* __restrict__ counts,
                                                         const int* __restrict__ base,
                                                         const int* __restrict__ tok_id) {
    const int e   = blockIdx.z;
    const int cnt = counts[e];
    const int t32 = blockIdx.x * 32;
    if (t32 >= cnt) return;

    const int wave = threadIdx.x >> 5;
    const int l    = threadIdx.x & 31;
    const int m    = l & 15;
    const int half = l >> 4;
    const int f0   = blockIdx.y * 256 + wave * 32;

    const int slot0 = t32 + m;
    const int slot1 = t32 + 16 + m;
    const int row0  = (slot0 < cnt) ? tok_id[e * T + slot0] : 0;
    const int row1  = (slot1 < cnt) ? tok_id[e * T + slot1] : 0;
    const unsigned short* a0row  = xb + (size_t)row0 * H;
    const unsigned short* a1row  = xb + (size_t)row1 * H;
    const unsigned short* b1row0 = w1b + ((size_t)e * F + (f0 + m)) * H;
    const unsigned short* b1row1 = w1b + ((size_t)e * F + (f0 + 16 + m)) * H;
    const unsigned short* b3row0 = w3b + ((size_t)e * F + (f0 + m)) * H;
    const unsigned short* b3row1 = w3b + ((size_t)e * F + (f0 + 16 + m)) * H;

    v8f acc1_00 = {}, acc1_01 = {}, acc1_10 = {}, acc1_11 = {};
    v8f acc3_00 = {}, acc3_01 = {}, acc3_10 = {}, acc3_11 = {};
    for (int k0 = 0; k0 < H; k0 += 32) {
        __builtin_prefetch(b1row0 + k0 + 64, 0, 3);
        __builtin_prefetch(b1row1 + k0 + 64, 0, 3);
        __builtin_prefetch(b3row0 + k0 + 64, 0, 3);
        __builtin_prefetch(b3row1 + k0 + 64, 0, 3);
        Frag16 a0, a1, b1f0, b1f1, b3f0, b3f1;
        load_a(a0, a0row, k0, half);
        load_a(a1, a1row, k0, half);
        load_b(b1f0, b1row0, k0, half);
        load_b(b1f1, b1row1, k0, half);
        load_b(b3f0, b3row0, k0, half);
        load_b(b3f1, b3row1, k0, half);
        acc1_00 = wmma_bf16(a0.v, b1f0.v, acc1_00);
        acc1_01 = wmma_bf16(a0.v, b1f1.v, acc1_01);
        acc1_10 = wmma_bf16(a1.v, b1f0.v, acc1_10);
        acc1_11 = wmma_bf16(a1.v, b1f1.v, acc1_11);
        acc3_00 = wmma_bf16(a0.v, b3f0.v, acc3_00);
        acc3_01 = wmma_bf16(a0.v, b3f1.v, acc3_01);
        acc3_10 = wmma_bf16(a1.v, b3f0.v, acc3_10);
        acc3_11 = wmma_bf16(a1.v, b3f1.v, acc3_11);
    }

    const int rb = base[e];
    const int n0 = f0 + m;
    const int n1 = f0 + 16 + m;
    #pragma unroll
    for (int r = 0; r < 8; ++r) {
        const int M = r + 8 * half;            // C layout: VGPR r -> rows M, M+8
        const int sA = t32 + M;
        if (sA < cnt) {
            unsigned short* grow = gated + (size_t)(rb + sA) * F;
            grow[n0] = f32_to_bf16_rne(silu_gate(acc1_00[r], acc3_00[r]));
            grow[n1] = f32_to_bf16_rne(silu_gate(acc1_01[r], acc3_01[r]));
        }
        const int sB = t32 + 16 + M;
        if (sB < cnt) {
            unsigned short* grow = gated + (size_t)(rb + sB) * F;
            grow[n0] = f32_to_bf16_rne(silu_gate(acc1_10[r], acc3_10[r]));
            grow[n1] = f32_to_bf16_rne(silu_gate(acc1_11[r], acc3_11[r]));
        }
    }
}

// ---------------- stage B: xout[row, h] = gated[row,:] @ w2[e,h,:]^T (f32 rows) ----------------
// wave tile: 32 rows x 32 h; grid: (T/32, H/256, E); block: 256 (8 waves)
__global__ __launch_bounds__(256) void moe_stage2_kernel(const unsigned short* __restrict__ gated,
                                                         const unsigned short* __restrict__ w2b,
                                                         float* __restrict__ xout,
                                                         const int* __restrict__ counts,
                                                         const int* __restrict__ base) {
    const int e   = blockIdx.z;
    const int cnt = counts[e];
    const int t32 = blockIdx.x * 32;
    if (t32 >= cnt) return;

    const int wave = threadIdx.x >> 5;
    const int l    = threadIdx.x & 31;
    const int m    = l & 15;
    const int half = l >> 4;
    const int h0   = blockIdx.y * 256 + wave * 32;
    const int rb   = base[e];

    const int slot0 = t32 + m;
    const int slot1 = t32 + 16 + m;
    const int rowA0 = rb + ((slot0 < cnt) ? slot0 : 0);   // pad -> replay row (discarded)
    const int rowA1 = rb + ((slot1 < cnt) ? slot1 : 0);
    const unsigned short* a0row = gated + (size_t)rowA0 * F;
    const unsigned short* a1row = gated + (size_t)rowA1 * F;
    const unsigned short* brow0 = w2b + ((size_t)e * H + (h0 + m)) * F;
    const unsigned short* brow1 = w2b + ((size_t)e * H + (h0 + 16 + m)) * F;

    v8f acc00 = {}, acc01 = {}, acc10 = {}, acc11 = {};
    for (int k0 = 0; k0 < F; k0 += 32) {
        __builtin_prefetch(brow0 + k0 + 64, 0, 3);
        __builtin_prefetch(brow1 + k0 + 64, 0, 3);
        Frag16 a0, a1, b0, b1;
        load_a(a0, a0row, k0, half);
        load_a(a1, a1row, k0, half);
        load_b(b0, brow0, k0, half);
        load_b(b1, brow1, k0, half);
        acc00 = wmma_bf16(a0.v, b0.v, acc00);
        acc01 = wmma_bf16(a0.v, b1.v, acc01);
        acc10 = wmma_bf16(a1.v, b0.v, acc10);
        acc11 = wmma_bf16(a1.v, b1.v, acc11);
    }

    const int n0 = h0 + m;
    const int n1 = h0 + 16 + m;
    #pragma unroll
    for (int r = 0; r < 8; ++r) {
        const int M = r + 8 * half;
        const int sA = t32 + M;
        if (sA < cnt) {
            float* orow = xout + (size_t)(rb + sA) * H;
            orow[n0] = acc00[r];
            orow[n1] = acc01[r];
        }
        const int sB = t32 + 16 + M;
        if (sB < cnt) {
            float* orow = xout + (size_t)(rb + sB) * H;
            orow[n0] = acc10[r];
            orow[n1] = acc11[r];
        }
    }
}

// ---------------- combine: out[t] = w0 * xout[row0] + w1 * xout[row1] (deterministic) ----------------
// grid: T blocks x 256 threads; each thread handles one float4 of H
__global__ __launch_bounds__(256) void combine_kernel(const float* __restrict__ xout,
                                                      const int* __restrict__ tok_e2,
                                                      const int* __restrict__ tok_pos2,
                                                      const float* __restrict__ tok_wt2,
                                                      const int* __restrict__ base,
                                                      float* __restrict__ out) {
    const int tok = blockIdx.x;
    const int i   = threadIdx.x;           // 0..255 -> float4 index over H=1024
    const int e0  = tok_e2[2 * tok];
    const int e1  = tok_e2[2 * tok + 1];
    const size_t row0 = (size_t)(base[e0] + tok_pos2[2 * tok]);
    const size_t row1 = (size_t)(base[e1] + tok_pos2[2 * tok + 1]);
    const float w0 = tok_wt2[2 * tok];
    const float w1 = tok_wt2[2 * tok + 1];

    const float4 r0 = ((const float4*)(xout + row0 * H))[i];
    const float4 r1 = ((const float4*)(xout + row1 * H))[i];
    float4 o;
    o.x = w0 * r0.x + w1 * r1.x;
    o.y = w0 * r0.y + w1 * r1.y;
    o.z = w0 * r0.z + w1 * r1.z;
    o.w = w0 * r0.w + w1 * r1.w;
    ((float4*)(out + (size_t)tok * H))[i] = o;
}

// ---------------- launch ----------------
extern "C" void kernel_launch(void* const* d_in, const int* in_sizes, int n_in,
                              void* d_out, int out_size, void* d_ws, size_t ws_size,
                              hipStream_t stream) {
    (void)in_sizes; (void)n_in; (void)out_size; (void)ws_size;

    const float* x  = (const float*)d_in[0];   // [T, H]
    const float* gw = (const float*)d_in[1];   // [E, H]
    const float* w1 = (const float*)d_in[2];   // [E, F, H]
    const float* w2 = (const float*)d_in[3];   // [E, H, F]
    const float* w3 = (const float*)d_in[4];   // [E, F, H]

    float* out_final  = (float*)d_out;                   // [T, H]
    float* out_logits = (float*)d_out + (size_t)T * H;   // [T, E]

    // workspace carve-up (256B aligned)
    char* ws = (char*)d_ws;
    size_t off = 0;
    auto carve = [&](size_t bytes) { char* p = ws + off; off = (off + bytes + 255) & ~(size_t)255; return p; };
    unsigned short* xb    = (unsigned short*)carve((size_t)T * H * 2);
    unsigned short* w1b   = (unsigned short*)carve((size_t)EFH * 2);
    unsigned short* w3b   = (unsigned short*)carve((size_t)EFH * 2);
    unsigned short* w2b   = (unsigned short*)carve((size_t)EFH * 2);
    unsigned short* gated = (unsigned short*)carve((size_t)NROWS * F * 2);
    float* xout   = (float*)carve((size_t)NROWS * H * sizeof(float));
    int*   counts = (int*)carve(E * sizeof(int));
    int*   rbase  = (int*)carve(E * sizeof(int));
    int*   tok_id = (int*)carve((size_t)E * T * sizeof(int));
    int*   tok_e2   = (int*)carve((size_t)2 * T * sizeof(int));
    int*   tok_pos2 = (int*)carve((size_t)2 * T * sizeof(int));
    float* tok_wt2  = (float*)carve((size_t)2 * T * sizeof(float));

    hipMemsetAsync(counts, 0, E * sizeof(int), stream);

    // f32 -> bf16 conversions
    cvt_bf16_kernel<<<4096, 256, 0, stream>>>(x,  xb,  T * H);
    cvt_bf16_kernel<<<16384, 256, 0, stream>>>(w1, w1b, EFH);
    cvt_bf16_kernel<<<16384, 256, 0, stream>>>(w3, w3b, EFH);
    cvt_bf16_kernel<<<16384, 256, 0, stream>>>(w2, w2b, EFH);

    // routing (exact f32) + gather lists + inverse map
    router_kernel<<<T, 128, 0, stream>>>(x, gw, out_logits, counts, tok_id,
                                         tok_e2, tok_pos2, tok_wt2);
    prefix_kernel<<<1, 32, 0, stream>>>(counts, rbase);

    // expert FFN: gated = silu(x@w1^T)*(x@w3^T)
    dim3 gridA(T / 32, F / 256, E);
    moe_stage1_kernel<<<gridA, 256, 0, stream>>>(xb, w1b, w3b, gated, counts, rbase, tok_id);

    // per-row output: xout = gated @ w2^T
    dim3 gridB(T / 32, H / 256, E);
    moe_stage2_kernel<<<gridB, 256, 0, stream>>>(gated, w2b, xout, counts, rbase);

    // deterministic 2-term combine
    combine_kernel<<<T, 256, 0, stream>>>(xout, tok_e2, tok_pos2, tok_wt2, rbase, out_final);
}
